// CompressedRelativeCausalAttention_17575006175527
// MI455X (gfx1250) — matrix operand order repaired and checked by reading, not compiled, not run on hardware
//
#include <hip/hip_runtime.h>
#include <math.h>

typedef __attribute__((ext_vector_type(16))) _Float16 v16h;
typedef __attribute__((ext_vector_type(8)))  _Float16 v8h;
typedef __attribute__((ext_vector_type(8)))  float    v8f;

#define S_  384
#define B_  2
#define C_  768
#define H_  12
#define CC_ 64
#define C3_ 2304

#define BM 64
#define BN 64
#define KT 64    // k-tile per LDS stage (two 32-wide WMMA substeps)
#define LDK 72   // padded LDS row length in halfs (144B rows -> conflict-free b128 frag loads)

// ---------------------------------------------------------------------------
// Generic batched WMMA GEMM, double-buffered LDS, fully specialized by template:
//   C[m,n] = sum_k (A[m,k] [+ addA[k]]) * Bsel[k,n]   [+ bias[n]]
//   Bsel[k,n] = TRANSB ? B[k*ldb + n] : B[n*ldb + k]
// CAUSAL==1: skip whole output tiles with min(t)=n0 > max(s)=m0+BM-1  (dd scores)
// CAUSAL==2: clamp K loop to k < m0+BM  (o = p @ v ; p[t>s]==0)
// All M,N multiples of 64; all (clamped) K multiples of 64.
// ---------------------------------------------------------------------------
template <int TRANSB, int CAUSAL, bool HASADD, bool HASBIAS>
__global__ __launch_bounds__(128)
void wmma_gemm(const float* __restrict__ A, const float* __restrict__ B,
               float* __restrict__ Cp, const float* __restrict__ bias,
               const float* __restrict__ addA,
               int M, int N, int K, int lda, int ldb, int ldc,
               int batchH,
               long sAo, long sAi, long sBo, long sBi, long sCo, long sCi,
               int addAi)
{
    __shared__ _Float16 As[2][BM][LDK];
    __shared__ _Float16 Bs[2][BN][LDK];

    const int bb = blockIdx.z;
    const int bo = bb / batchH, bi = bb % batchH;
    const float* Ab = A + (long)bo * sAo + (long)bi * sAi;
    const float* Bb = B + (long)bo * sBo + (long)bi * sBi;
    float*       Cb = Cp + (long)bo * sCo + (long)bi * sCi;
    const float* uAb = HASADD ? (addA + bi * addAi) : nullptr;

    const int m0 = blockIdx.y * BM;
    const int n0 = blockIdx.x * BN;
    if (CAUSAL == 1 && n0 > m0 + (BM - 1)) return;   // fully-masked tile

    const int tid  = threadIdx.x;
    const int lane = tid & 31;
    const int wid  = tid >> 5;
    const int wm   = (wid >> 1) * 32;   // wave tile origin in block tile
    const int wn   = (wid & 1)  * 32;
    const int lr   = lane & 15;         // frag row/col
    const int lg   = lane >> 4;         // frag half-group

    const int kk   = (tid & 15) * 4;    // staging column  (0..60)
    const int rw0  = tid >> 4;          // staging row base (0..7), step 8

    v8f acc[2][2] = {};

    const int kend   = (CAUSAL == 2) ? ((m0 + BM) < K ? (m0 + BM) : K) : K;
    const int ktiles = kend / KT;

    float4 ra[8], rb[8];

    auto loadA = [&](int k0) {
        #pragma unroll
        for (int r = 0; r < 8; ++r) {
            const int row = rw0 + 8 * r;
            float4 av = *(const float4*)(Ab + (long)(m0 + row) * lda + (k0 + kk));
            if (HASADD) {
                av.x += uAb[k0 + kk + 0]; av.y += uAb[k0 + kk + 1];
                av.z += uAb[k0 + kk + 2]; av.w += uAb[k0 + kk + 3];
            }
            ra[r] = av;
        }
    };
    auto loadB = [&](int k0) {
        #pragma unroll
        for (int r = 0; r < 8; ++r) {
            const int row = rw0 + 8 * r;   // n-row (NT) or k-row (NN)
            rb[r] = TRANSB
                ? *(const float4*)(Bb + (long)(k0 + row) * ldb + (n0 + kk))
                : *(const float4*)(Bb + (long)(n0 + row) * ldb + (k0 + kk));
        }
    };
    auto storeAB = [&](int buf) {
        #pragma unroll
        for (int r = 0; r < 8; ++r) {
            const int row = rw0 + 8 * r;
            As[buf][row][kk + 0] = (_Float16)ra[r].x;
            As[buf][row][kk + 1] = (_Float16)ra[r].y;
            As[buf][row][kk + 2] = (_Float16)ra[r].z;
            As[buf][row][kk + 3] = (_Float16)ra[r].w;
            if (!TRANSB) {
                Bs[buf][row][kk + 0] = (_Float16)rb[r].x;
                Bs[buf][row][kk + 1] = (_Float16)rb[r].y;
                Bs[buf][row][kk + 2] = (_Float16)rb[r].z;
                Bs[buf][row][kk + 3] = (_Float16)rb[r].w;
            } else {           // transpose into Bs[n][k]
                Bs[buf][kk + 0][row] = (_Float16)rb[r].x;
                Bs[buf][kk + 1][row] = (_Float16)rb[r].y;
                Bs[buf][kk + 2][row] = (_Float16)rb[r].z;
                Bs[buf][kk + 3][row] = (_Float16)rb[r].w;
            }
        }
    };
    auto compute = [&](int buf) {
        #pragma unroll
        for (int ks = 0; ks < 2; ++ks) {         // two 32-wide K substeps
            v16h af[2], bf[2];
            #pragma unroll
            for (int i = 0; i < 2; ++i) {
                // A 16x32 frag: lane lr = row M; element j -> K = 16*(j/8)+8*lg+(j%8)
                const _Float16* ar = &As[buf][wm + 16 * i + lr][32 * ks];
                v8h alo = *(const v8h*)(ar + 8 * lg);
                v8h ahi = *(const v8h*)(ar + 16 + 8 * lg);
                #pragma unroll
                for (int j = 0; j < 8; ++j) { af[i][j] = alo[j]; af[i][8 + j] = ahi[j]; }
                // B 32x16 frag: lane lr = col N; element j -> K = 16*(j/8)+8*lg+(j%8)
                const _Float16* br = &Bs[buf][wn + 16 * i + lr][32 * ks];
                v8h blo = *(const v8h*)(br + 8 * lg);
                v8h bhi = *(const v8h*)(br + 16 + 8 * lg);
                #pragma unroll
                for (int j = 0; j < 8; ++j) { bf[i][j] = blo[j]; bf[i][8 + j] = bhi[j]; }
            }
            #pragma unroll
            for (int im = 0; im < 2; ++im)
                #pragma unroll
                for (int in = 0; in < 2; ++in)
                    acc[im][in] = __builtin_amdgcn_wmma_f32_16x16x32_f16(
                        false, af[im], false, bf[in], (short)0, acc[im][in], false, false);
        }
    };

    // prologue: stage tile 0
    loadA(0); loadB(0);
    storeAB(0);
    __syncthreads();

    for (int kt = 0; kt < ktiles; ++kt) {
        const int  cur  = kt & 1;
        const bool more = (kt + 1) < ktiles;
        if (more) { loadA((kt + 1) * KT); loadB((kt + 1) * KT); }  // issue global loads early
        compute(cur);                                              // WMMAs on current buffer
        if (more) storeAB(cur ^ 1);                                // fill next buffer
        __syncthreads();
    }

    // store: C/D frag: VGPR r -> M = r (lanes 0-15) / 8+r (lanes 16-31), N = lr
    #pragma unroll
    for (int im = 0; im < 2; ++im) {
        #pragma unroll
        for (int in = 0; in < 2; ++in) {
            const int nn = n0 + wn + 16 * in + lr;
            #pragma unroll
            for (int r = 0; r < 8; ++r) {
                const int mm = m0 + wm + 16 * im + lg * 8 + r;
                float v = acc[im][in][r];
                if (HASBIAS) v += bias[nn];
                Cb[(long)mm * ldc + nn] = v;
            }
        }
    }
}

// ---------------------------------------------------------------------------
// dp_wp: scores[b,h,s,t] += dot(q[s,b,h,:]+w[h,:], pe[s,t,b,h,:])  for t <= s
// One block per (s,b); 8 waves; wave handles t = wid, wid+8, ...
// Streams only the causal half of pe (halves the 906 MB dominant read).
// ---------------------------------------------------------------------------
__global__ __launch_bounds__(256)
void dp_kernel(const float* __restrict__ qkv, const float* __restrict__ pe,
               const float* __restrict__ w, float* __restrict__ scores)
{
    const int s = blockIdx.x, b = blockIdx.y;
    const int tid = threadIdx.x, wid = tid >> 5, lane = tid & 31;
    const int grp = lane >> 4;

    // preload q+w for this (s,b): lane covers c = 4*lane + 128*i, i=0..5
    float4 qw[6];
    const float* qrow = qkv + (long)(s * B_ + b) * C3_;
    #pragma unroll
    for (int i = 0; i < 6; ++i) {
        const int c = 4 * lane + 128 * i;
        float4 qv = *(const float4*)(qrow + c);
        float4 wv = *(const float4*)(w + c);
        qw[i] = make_float4(qv.x + wv.x, qv.y + wv.y, qv.z + wv.z, qv.w + wv.w);
    }

    for (int t = wid; t <= s; t += 8) {
        const float* perow = pe + (((long)s * S_ + t) * B_ + b) * C_;

        // prefetch the row this wave touches next iteration (global_prefetch_b8)
        if (t + 8 <= s) {
            const float* nrow = pe + (((long)s * S_ + (t + 8)) * B_ + b) * C_;
            __builtin_prefetch(nrow + 4 * lane, 0, 1);
            __builtin_prefetch(nrow + 4 * lane + 384, 0, 1);
        }

        // issue all 6 b128 loads up front, then overlap reductions with latency
        float4 p4[6];
        #pragma unroll
        for (int i = 0; i < 6; ++i)
            p4[i] = *(const float4*)(perow + 4 * lane + 128 * i);

        #pragma unroll
        for (int i = 0; i < 6; ++i) {
            float acc = p4[i].x * qw[i].x + p4[i].y * qw[i].y
                      + p4[i].z * qw[i].z + p4[i].w * qw[i].w;
            // reduce within each 16-lane half (h = 2i for lanes 0-15, 2i+1 for 16-31)
            acc += __shfl_xor(acc, 1);
            acc += __shfl_xor(acc, 2);
            acc += __shfl_xor(acc, 4);
            acc += __shfl_xor(acc, 8);
            if ((lane & 15) == 0) {
                const int h = 2 * i + grp;
                scores[(((long)b * H_ + h) * S_ + s) * S_ + t] += acc;
            }
        }
    }
}

// ---------------------------------------------------------------------------
// Causal softmax over t, in place: scores -> p. One wave per (b,h,s) row.
// Writes zeros for t > s so downstream p@v can read full rows.
// ---------------------------------------------------------------------------
__global__ __launch_bounds__(128)
void softmax_kernel(float* __restrict__ scores)
{
    const int row  = blockIdx.x * 4 + (threadIdx.x >> 5);   // over B*H*S
    const int lane = threadIdx.x & 31;
    const int s = row % S_;
    float* r = scores + (long)row * S_;
    const int n = s + 1;
    const float scale = 0.125f;  // 1/sqrt(64)

    float vals[12];
    float m = -INFINITY;
    #pragma unroll
    for (int i = 0; i < 12; ++i) {
        const int t = lane + 32 * i;
        float v = (t < n) ? r[t] * scale : -INFINITY;
        vals[i] = v;
        m = fmaxf(m, v);
    }
    #pragma unroll
    for (int msk = 16; msk >= 1; msk >>= 1) m = fmaxf(m, __shfl_xor(m, msk));

    float sum = 0.f;
    #pragma unroll
    for (int i = 0; i < 12; ++i) {
        const int t = lane + 32 * i;
        float e = (t < n) ? __expf(vals[i] - m) : 0.f;
        vals[i] = e;
        sum += e;
    }
    #pragma unroll
    for (int msk = 16; msk >= 1; msk >>= 1) sum += __shfl_xor(sum, msk);
    const float inv = 1.f / sum;

    #pragma unroll
    for (int i = 0; i < 12; ++i) {
        const int t = lane + 32 * i;
        if (t < S_) r[t] = vals[i] * inv;
    }
}

// ---------------------------------------------------------------------------
extern "C" void kernel_launch(void* const* d_in, const int* in_sizes, int n_in,
                              void* d_out, int out_size, void* d_ws, size_t ws_size,
                              hipStream_t stream)
{
    (void)in_sizes; (void)n_in; (void)out_size; (void)ws_size;
    const float* x     = (const float*)d_in[0];
    const float* pe    = (const float*)d_in[1];
    // d_in[2]=cm, d_in[3]=pm: causality handled analytically
    const float* qkv_w = (const float*)d_in[4];
    const float* qkv_b = (const float*)d_in[5];
    const float* u     = (const float*)d_in[6];
    const float* w     = (const float*)d_in[7];
    const float* out_w = (const float*)d_in[8];
    const float* out_b = (const float*)d_in[9];
    float* out = (float*)d_out;

    float* qkv    = (float*)d_ws;                         // (S,B,3C)  7.1 MB
    float* scores = qkv + (long)S_ * B_ * C3_;            // (B,H,S,S) 14.2 MB (reused as p)
    float* obuf   = scores + (long)B_ * H_ * S_ * S_;     // (S,B,C)   2.4 MB

    // 1) qkv = x @ qkv_w.T + qkv_b : M=768, N=2304, K=768 (NT, bias)
    wmma_gemm<0, 0, false, true><<<dim3(C3_ / BN, (S_ * B_) / BM, 1), 128, 0, stream>>>(
        x, qkv_w, qkv, qkv_b, nullptr,
        S_ * B_, C3_, C_, C_, C_, C3_,
        1, 0, 0, 0, 0, 0, 0, 0);

    // 2) dd_ud: scores[b,h,s,t] = (q+u) . k : batched (b,h), NT, u-add, causal tile skip
    wmma_gemm<0, 1, true, false><<<dim3(S_ / BN, S_ / BM, B_ * H_), 128, 0, stream>>>(
        qkv, qkv + C_, scores, nullptr, u,
        S_, S_, CC_, B_ * C3_, B_ * C3_, S_,
        H_, C3_, CC_, C3_, CC_,
        (long)H_ * S_ * S_, (long)S_ * S_, CC_);

    // 3) dp_wp: stream causal half of pe, add into scores
    dp_kernel<<<dim3(S_, B_), 256, 0, stream>>>(qkv, pe, w, scores);

    // 4) causal softmax in place (scores -> p)
    softmax_kernel<<<(B_ * H_ * S_) / 4, 128, 0, stream>>>(scores);

    // 5) o = p @ v : batched (b,h), NN, causal k-clamp
    wmma_gemm<1, 2, false, false><<<dim3(CC_ / BN, S_ / BM, B_ * H_), 128, 0, stream>>>(
        scores, qkv + 2 * C_, obuf, nullptr, nullptr,
        S_, CC_, S_, S_, B_ * C3_, B_ * C_,
        H_, (long)H_ * S_ * S_, (long)S_ * S_, C3_, CC_,
        (long)C_, (long)CC_, 0);

    // 6) out = o @ out_w.T + out_b : M=768, N=768, K=768 (NT, bias)
    wmma_gemm<0, 0, false, true><<<dim3(C_ / BN, (S_ * B_) / BM, 1), 128, 0, stream>>>(
        obuf, out_w, out, out_b, nullptr,
        S_ * B_, C_, C_, C_, C_, C_,
        1, 0, 0, 0, 0, 0, 0, 0);
}
